// MultiHeadSelfAttention_69887707840816
// MI455X (gfx1250) — matrix operand order repaired and checked by reading, not compiled, not run on hardware
//
#include <hip/hip_runtime.h>
#include <hip/hip_bf16.h>

#define NHEAD    16
#define HEAD_DIM 64
#define DMODEL   1024
#define INNER    1024
#define BATCH    2
#define SEQ      2048
#define BT       (BATCH * SEQ)

typedef __attribute__((ext_vector_type(16))) __bf16 v16bf;
typedef __attribute__((ext_vector_type(8)))  float  v8f;
typedef unsigned int u32x4 __attribute__((ext_vector_type(4)));
typedef unsigned int u32x8 __attribute__((ext_vector_type(8)));

union Frag16 { v16bf v; unsigned short u[16]; };

struct alignas(8)  US4 { unsigned short a, b, c, d; };
struct alignas(16) UI4 { unsigned int a, b, c, d; };

// ---------- helpers ----------

__device__ __forceinline__ unsigned short f2bf(float f) {
  unsigned int u = __builtin_bit_cast(unsigned int, f);
  u += 0x7fffu + ((u >> 16) & 1u);               // round-to-nearest-even
  return (unsigned short)(u >> 16);
}

// A fragment, 16x32 bf16 (M x K). base points at [16][ld] (row-major bf16, 16B-aligned rows).
// Layout: lanes 0-15 row M=lane hold K {0..15}; lanes 16-31 hold K {16..31} (matches B loaders).
__device__ __forceinline__ Frag16 load_a16x32(const unsigned short* base, int ld, int lane) {
  Frag16 f;
  const int r = lane & 15, h = (lane >> 4) & 1;
#pragma unroll
  for (int p = 0; p < 8; ++p) {
    const int kb = 16 * h + 2 * p;
    f.u[2 * p]     = base[r * ld + kb];
    f.u[2 * p + 1] = base[r * ld + kb + 1];
  }
  return f;
}

// B fragment, 32x16 bf16 (K x N), tile stored [n][k] (row = output column).
__device__ __forceinline__ Frag16 load_b32x16_nk(const unsigned short* base, int ld, int lane) {
  Frag16 f;
  const int n = lane & 15, kh = (lane >> 4) & 1;
#pragma unroll
  for (int p = 0; p < 8; ++p) {
    const int kb = 16 * kh + 2 * p;
    f.u[2 * p]     = base[n * ld + kb];
    f.u[2 * p + 1] = base[n * ld + kb + 1];
  }
  return f;
}

// B fragment, 32x16 bf16 (K x N), tile stored [k][n].
__device__ __forceinline__ Frag16 load_b32x16_kn(const unsigned short* base, int ld, int lane) {
  Frag16 f;
  const int n = lane & 15, kh = (lane >> 4) & 1;
#pragma unroll
  for (int p = 0; p < 8; ++p) {
    const int kb = 16 * kh + 2 * p;
    f.u[2 * p]     = base[kb * ld + n];
    f.u[2 * p + 1] = base[(kb + 1) * ld + n];
  }
  return f;
}

__device__ __forceinline__ v8f wmma_bf16(const Frag16& a, const Frag16& b, v8f c) {
  return __builtin_amdgcn_wmma_f32_16x16x32_bf16(false, a.v, false, b.v, (short)0, c,
                                                 false, false);
}

// ds_swizzle xor within 16-lane groups (and_mask=0x1f keeps it in-wave).
#define SWZ(x, mask)                                                                        \
  __builtin_bit_cast(float, __builtin_amdgcn_ds_swizzle(__builtin_bit_cast(int, (x)),       \
                                                        0x1f | ((mask) << 10)))

__device__ __forceinline__ float redmax16(float x) {
  x = fmaxf(x, SWZ(x, 1));
  x = fmaxf(x, SWZ(x, 2));
  x = fmaxf(x, SWZ(x, 4));
  x = fmaxf(x, SWZ(x, 8));
  return x;
}
__device__ __forceinline__ float redsum16(float x) {
  x += SWZ(x, 1);
  x += SWZ(x, 2);
  x += SWZ(x, 4);
  x += SWZ(x, 8);
  return x;
}

// LDS row stride for 64-wide bf16 tiles: 72 ushorts = 144 B (multiple of 16 -> b128-friendly).
#define LDT 72

// Issue one TDM 2D tile load: 32 rows x 64 bf16, row stride 64 elements in memory,
// +16B LDS padding after every 128B row (keeps the 144B LDS stride). D# per ISA 8.3/8.4.
__device__ __forceinline__ void tdm_load_tile_32x64(unsigned lds_addr, const void* gptr) {
  const unsigned long long ga = (unsigned long long)(uintptr_t)gptr;
  u32x4 g0;
  g0[0] = 1u;                                           // count=1, user descriptor
  g0[1] = lds_addr;                                     // LDS byte address
  g0[2] = (unsigned)ga;                                 // global_addr[31:0]
  g0[3] = ((unsigned)(ga >> 32) & 0x01FFFFFFu) | (2u << 30);  // addr[56:32] | type=2
  u32x8 g1;
  g1[0] = (1u << 16)        // data_size = 2 bytes
        | (1u << 20)        // pad_enable
        | (4u << 22)        // pad_interval: 32 DWORDs (=128B row)
        | (3u << 25);       // pad_amount: 4 DWORDs (=16B)
  g1[1] = (64u << 16);      // tensor_dim0 = 64 (bits 79:48, low half)
  g1[2] = (32u << 16);      // tensor_dim1 = 32 (bits 111:80, low half)
  g1[3] = (64u << 16);      // tile_dim0 = 64 (bits 127:112)
  g1[4] = 32u;              // tile_dim1 = 32 (bits 143:128)
  g1[5] = 64u;              // tensor_dim0_stride = 64 elements (bits 207:160, low)
  g1[6] = 0u;
  g1[7] = 0u;
  asm volatile("tensor_load_to_lds %0, %1" :: "s"(g0), "s"(g1) : "memory");
}

// stage a 128x64 f32->bf16 tile into LDS ([row][k] for A, [n][k] for B); 256 threads.
__device__ __forceinline__ void stage_f32_tile_128x64(unsigned short (*dst)[LDT],
                                                      const float* __restrict__ src,
                                                      int row0, size_t ldsrc, int k0, int tid) {
#pragma unroll
  for (int i = 0; i < 8; ++i) {
    const int e = tid + 256 * i;               // 0..2047 float4s
    const int r = e >> 4, c4 = e & 15;
    const float4 f = *(const float4*)(&src[(size_t)(row0 + r) * ldsrc + k0 + c4 * 4]);
    US4 u{f2bf(f.x), f2bf(f.y), f2bf(f.z), f2bf(f.w)};
    *(US4*)(&dst[r][c4 * 4]) = u;
  }
}

// ---------- kernel 1: fused QKV projection  (x @ Wqkv^T + b) -> Q/K/V bf16 [B,H,T,64] ----------

__global__ __launch_bounds__(256) void qkv_gemm_kernel(
    const float* __restrict__ x, const float* __restrict__ qkv_w,
    const float* __restrict__ qkv_b, unsigned short* __restrict__ q_ws,
    unsigned short* __restrict__ k_ws, unsigned short* __restrict__ v_ws) {
  __shared__ unsigned short As[128][LDT];
  __shared__ unsigned short Bs[128][LDT];

  const int tid = threadIdx.x;
  const int lane = tid & 31, wave = tid >> 5;
  const int wm = wave >> 2, wn = wave & 3;           // 2x4 wave grid -> 128x128 block tile
  const int M0 = blockIdx.x * 128;
  const int N0 = blockIdx.y * 128;

  v8f acc[4][2];
#pragma unroll
  for (int i = 0; i < 4; ++i)
#pragma unroll
    for (int j = 0; j < 2; ++j) acc[i][j] = v8f{0, 0, 0, 0, 0, 0, 0, 0};

  for (int k0 = 0; k0 < DMODEL; k0 += 64) {
    if (k0 + 64 < DMODEL) {
      __builtin_prefetch(&x[(size_t)(M0 + (tid >> 1)) * DMODEL + k0 + 64], 0, 1);
      __builtin_prefetch(&qkv_w[(size_t)(N0 + (tid >> 1)) * DMODEL + k0 + 64], 0, 1);
    }
    stage_f32_tile_128x64(As, x, M0, DMODEL, k0, tid);
    stage_f32_tile_128x64(Bs, qkv_w, N0, DMODEL, k0, tid);
    __syncthreads();

#pragma unroll
    for (int ks = 0; ks < 64; ks += 32) {            // two K=32 sub-steps per stage
      Frag16 a[4], b[2];
#pragma unroll
      for (int i = 0; i < 4; ++i) a[i] = load_a16x32(&As[wm * 64 + 16 * i][ks], LDT, lane);
#pragma unroll
      for (int j = 0; j < 2; ++j) b[j] = load_b32x16_nk(&Bs[wn * 32 + 16 * j][ks], LDT, lane);
#pragma unroll
      for (int i = 0; i < 4; ++i)
#pragma unroll
        for (int j = 0; j < 2; ++j) acc[i][j] = wmma_bf16(a[i], b[j], acc[i][j]);
    }
    __syncthreads();
  }

  const int nlo = lane & 15, half = lane >> 4;
#pragma unroll
  for (int i = 0; i < 4; ++i)
#pragma unroll
    for (int j = 0; j < 2; ++j)
#pragma unroll
      for (int v = 0; v < 8; ++v) {
        const int row = M0 + wm * 64 + i * 16 + v + 8 * half;  // token index
        const int e   = N0 + wn * 32 + j * 16 + nlo;           // qkv feature index
        float val = acc[i][j][v] + qkv_b[e];
        const int s = e >> 10;                                 // 0=Q 1=K 2=V
        const int inner = e & 1023;
        const int h = inner >> 6, d = inner & 63;
        const int b = row >> 11, t = row & 2047;
        const size_t dst = (((size_t)(b * NHEAD + h)) * SEQ + t) * HEAD_DIM + d;
        if (s == 0)      q_ws[dst] = f2bf(val * 0.125f);       // fold 1/sqrt(64)
        else if (s == 1) k_ws[dst] = f2bf(val);
        else             v_ws[dst] = f2bf(val);
      }
}

// ---------- kernel 2: streaming softmax attention (mask keeps key >= query) ----------
// K/V chunks are double-buffered in LDS via the Tensor Data Mover: wave 0 issues the
// next chunk's descriptors before computing the current one, so DMA latency hides
// under the 8 WMMA + online-softmax of the resident chunk.

__global__ __launch_bounds__(128) void attn_kernel(
    const unsigned short* __restrict__ q_ws, const unsigned short* __restrict__ k_ws,
    const unsigned short* __restrict__ v_ws, unsigned short* __restrict__ ctx) {
  __shared__ unsigned short Ks[2][32][LDT];
  __shared__ unsigned short Vs[2][32][LDT];
  __shared__ unsigned short Ps[4][16][40];           // 80B rows: 16B aligned

  const int tid = threadIdx.x;
  const int lane = tid & 31, w = tid >> 5;
  const int bh = blockIdx.x;                         // b*16 + h
  const int Q0 = blockIdx.y * 64;
  const int qrow0 = Q0 + 16 * w;                     // this wave's 16 query rows
  const int nlo = lane & 15, half = lane >> 4;

  const unsigned short* kbase = k_ws + (size_t)bh * SEQ * HEAD_DIM;
  const unsigned short* vbase = v_ws + (size_t)bh * SEQ * HEAD_DIM;

  // Q fragments (pre-scaled), kept in registers for the whole key stream.
  const unsigned short* qbase = q_ws + ((size_t)bh * SEQ + qrow0) * HEAD_DIM;
  const Frag16 aQ0 = load_a16x32(qbase, HEAD_DIM, lane);       // d 0..31
  const Frag16 aQ1 = load_a16x32(qbase + 32, HEAD_DIM, lane);  // d 32..63

  v8f o[4];
#pragma unroll
  for (int dt = 0; dt < 4; ++dt) o[dt] = v8f{0, 0, 0, 0, 0, 0, 0, 0};
  float m[8], l[8];
#pragma unroll
  for (int v = 0; v < 8; ++v) { m[v] = -1e30f; l[v] = 0.0f; }

  const int nchunks = (SEQ - Q0) / 32;               // keys stream upward from Q0

  if (w == 0) {                                      // prologue: DMA chunk 0 into buffer 0
    tdm_load_tile_32x64((unsigned)(uintptr_t)&Ks[0][0][0], kbase + (size_t)Q0 * HEAD_DIM);
    tdm_load_tile_32x64((unsigned)(uintptr_t)&Vs[0][0][0], vbase + (size_t)Q0 * HEAD_DIM);
  }

  for (int c = 0; c < nchunks; ++c) {
    const int key0 = Q0 + 32 * c;
    const int buf = c & 1;
    if (w == 0) {
      if (c + 1 < nchunks) {                         // kick off next chunk, then wait for
        const size_t nxt = (size_t)(key0 + 32) * HEAD_DIM;   // current (<=2 outstanding)
        tdm_load_tile_32x64((unsigned)(uintptr_t)&Ks[buf ^ 1][0][0], kbase + nxt);
        tdm_load_tile_32x64((unsigned)(uintptr_t)&Vs[buf ^ 1][0][0], vbase + nxt);
        __builtin_amdgcn_s_wait_tensorcnt(2);
      } else {
        __builtin_amdgcn_s_wait_tensorcnt(0);
      }
    }
    __syncthreads();
    const unsigned short (*K)[LDT] = Ks[buf];
    const unsigned short (*V)[LDT] = Vs[buf];

    // S = Q . K^T for 16 queries x 32 keys (two 16x16 tiles, d-contraction in 2 steps)
    v8f s0 = v8f{0, 0, 0, 0, 0, 0, 0, 0};
    v8f s1 = v8f{0, 0, 0, 0, 0, 0, 0, 0};
    {
      Frag16 bK = load_b32x16_nk(&K[0][0], LDT, lane);
      s0 = wmma_bf16(aQ0, bK, s0);
      bK = load_b32x16_nk(&K[0][32], LDT, lane);
      s0 = wmma_bf16(aQ1, bK, s0);
      bK = load_b32x16_nk(&K[16][0], LDT, lane);
      s1 = wmma_bf16(aQ0, bK, s1);
      bK = load_b32x16_nk(&K[16][32], LDT, lane);
      s1 = wmma_bf16(aQ1, bK, s1);
    }

    // mask + online softmax (rows v + 8*half live across the 16-lane half-wave)
    float alpha[8];
#pragma unroll
    for (int v = 0; v < 8; ++v) {
      const int qr = qrow0 + v + 8 * half;
      if (key0 + nlo < qr)      s0[v] = -1e30f;
      if (key0 + 16 + nlo < qr) s1[v] = -1e30f;
      float mx = redmax16(fmaxf(s0[v], s1[v]));
      mx = fmaxf(mx, m[v]);
      alpha[v] = __expf(m[v] - mx);
      const float p0 = __expf(s0[v] - mx);
      const float p1 = __expf(s1[v] - mx);
      l[v] = l[v] * alpha[v] + redsum16(p0 + p1);
      m[v] = mx;
      const int r = v + 8 * half;
      Ps[w][r][nlo]      = f2bf(p0);                 // C-layout -> row-major P in LDS
      Ps[w][r][nlo + 16] = f2bf(p1);
    }
#pragma unroll
    for (int dt = 0; dt < 4; ++dt)
#pragma unroll
      for (int v = 0; v < 8; ++v) o[dt][v] *= alpha[v];

    asm volatile("s_wait_dscnt 0" ::: "memory");     // P stores visible to our reload
    const Frag16 aP = load_a16x32(&Ps[w][0][0], 40, lane);
#pragma unroll
    for (int dt = 0; dt < 4; ++dt) {                 // O += P . V   (K-dim = 32 keys)
      const Frag16 bV = load_b32x16_kn(&V[0][dt * 16], LDT, lane);
      o[dt] = wmma_bf16(aP, bV, o[dt]);
    }
    __syncthreads();                                 // all reads of buf done before reuse
  }

  // finalize: O / l -> ctx [B*T, INNER] bf16 with head-concat layout
  const int b = bh >> 4, h = bh & 15;
#pragma unroll
  for (int dt = 0; dt < 4; ++dt)
#pragma unroll
    for (int v = 0; v < 8; ++v) {
      const int t = qrow0 + v + 8 * half;
      const float val = o[dt][v] / l[v];
      ctx[((size_t)(b * SEQ) + t) * INNER + h * HEAD_DIM + dt * 16 + nlo] = f2bf(val);
    }
}

// ---------- kernel 3: output projection  (ctx @ Wout^T + b) -> fp32 ----------

__global__ __launch_bounds__(256) void out_gemm_kernel(
    const unsigned short* __restrict__ ctx, const float* __restrict__ out_w,
    const float* __restrict__ out_b, float* __restrict__ out) {
  __shared__ unsigned short As[128][LDT];
  __shared__ unsigned short Bs[128][LDT];

  const int tid = threadIdx.x;
  const int lane = tid & 31, wave = tid >> 5;
  const int wm = wave >> 2, wn = wave & 3;
  const int M0 = blockIdx.x * 128;
  const int N0 = blockIdx.y * 128;

  v8f acc[4][2];
#pragma unroll
  for (int i = 0; i < 4; ++i)
#pragma unroll
    for (int j = 0; j < 2; ++j) acc[i][j] = v8f{0, 0, 0, 0, 0, 0, 0, 0};

  for (int k0 = 0; k0 < INNER; k0 += 64) {
    if (k0 + 64 < INNER) {
      __builtin_prefetch(&ctx[(size_t)(M0 + (tid >> 1)) * INNER + k0 + 64], 0, 1);
      __builtin_prefetch(&out_w[(size_t)(N0 + (tid >> 1)) * INNER + k0 + 64], 0, 1);
    }
#pragma unroll
    for (int i = 0; i < 4; ++i) {                    // ctx already bf16: 16B copies
      const int e = tid + 256 * i;                   // 0..1023 uint4s
      const int r = e >> 3, c = e & 7;
      *(UI4*)&As[r][c * 8] = ((const UI4*)(ctx + (size_t)(M0 + r) * INNER + k0))[c];
    }
    stage_f32_tile_128x64(Bs, out_w, N0, INNER, k0, tid);
    __syncthreads();

#pragma unroll
    for (int ks = 0; ks < 64; ks += 32) {
      Frag16 a[4], b[2];
#pragma unroll
      for (int i = 0; i < 4; ++i) a[i] = load_a16x32(&As[wm * 64 + 16 * i][ks], LDT, lane);
#pragma unroll
      for (int j = 0; j < 2; ++j) b[j] = load_b32x16_nk(&Bs[wn * 32 + 16 * j][ks], LDT, lane);
#pragma unroll
      for (int i = 0; i < 4; ++i)
#pragma unroll
        for (int j = 0; j < 2; ++j) acc[i][j] = wmma_bf16(a[i], b[j], acc[i][j]);
    }
    __syncthreads();
  }

  const int nlo = lane & 15, half = lane >> 4;
#pragma unroll
  for (int i = 0; i < 4; ++i)
#pragma unroll
    for (int j = 0; j < 2; ++j)
#pragma unroll
      for (int v = 0; v < 8; ++v) {
        const int row = M0 + wm * 64 + i * 16 + v + 8 * half;
        const int e   = N0 + wn * 32 + j * 16 + nlo;
        out[(size_t)row * DMODEL + e] = acc[i][j][v] + out_b[e];
      }
}

// ---------- launch ----------

extern "C" void kernel_launch(void* const* d_in, const int* in_sizes, int n_in,
                              void* d_out, int out_size, void* d_ws, size_t ws_size,
                              hipStream_t stream) {
  (void)in_sizes; (void)n_in; (void)out_size; (void)ws_size;
  const float* x     = (const float*)d_in[0];
  const float* qkv_w = (const float*)d_in[1];
  const float* qkv_b = (const float*)d_in[2];
  const float* out_w = (const float*)d_in[3];
  const float* out_b = (const float*)d_in[4];
  float* out = (float*)d_out;

  const size_t per_tensor = (size_t)BATCH * NHEAD * SEQ * HEAD_DIM;  // 4M bf16
  unsigned short* q_ws = (unsigned short*)d_ws;
  unsigned short* k_ws = q_ws + per_tensor;
  unsigned short* v_ws = k_ws + per_tensor;
  unsigned short* ctx  = v_ws + per_tensor;                          // 4M bf16

  qkv_gemm_kernel<<<dim3(BT / 128, (3 * INNER) / 128), 256, 0, stream>>>(
      x, qkv_w, qkv_b, q_ws, k_ws, v_ws);
  attn_kernel<<<dim3(BATCH * NHEAD, SEQ / 64), 128, 0, stream>>>(q_ws, k_ws, v_ws, ctx);
  out_gemm_kernel<<<dim3(BT / 128, DMODEL / 128), 256, 0, stream>>>(ctx, out_w, out_b, out);
}
